// MSDeformMatchV3Attn_10136122818978
// MI455X (gfx1250) — matrix operand order repaired and checked by reading, not compiled, not run on hardware
//
#include <hip/hip_runtime.h>

typedef __bf16 v16bf __attribute__((ext_vector_type(16)));
typedef float  v8f   __attribute__((ext_vector_type(8)));

#define NB   2
#define TT   4
#define HS   32
#define WSZ  32
#define DD   128
#define LEN  4096            // TT*HS*WSZ
#define P9A  36
#define OUT_OFF (NB*LEN*DD)  // 1048576 floats: start of sampling_locations

// ---------------------------------------------------------------------------
// Prepass: input_flatten f32 -> bf16 (L2-resident operand for async staging)
// ---------------------------------------------------------------------------
__global__ void cvt_bf16(const float* __restrict__ src, __bf16* __restrict__ dst,
                         int ntot) {
  int i = blockIdx.x * 256 + threadIdx.x;
  if (i < ntot) dst[i] = (__bf16)src[i];
}

// ---------------------------------------------------------------------------
// Main fused kernel: async-staged ring buffer + bf16 WMMA
// ---------------------------------------------------------------------------
__launch_bounds__(128)
__global__ void msdma_fused_async(const float* __restrict__ query,
                                  const float* __restrict__ vflat,
                                  const __bf16* __restrict__ Vb,
                                  const float* __restrict__ Wout,
                                  const float* __restrict__ bout,
                                  float* __restrict__ out)
{
  __shared__ __bf16 Qs[3][34][128];            // 26112 B (reused as accbuf)
  __shared__ __bf16 Vs[4][34][128];            // 34816 B, rotating row slabs
  __shared__ __align__(16) char sbuf[4096];    // corrbuf / sridx union
  float (*corrbuf)[32] = (float (*)[32])sbuf;  // 32*32*4  = 4096 B
  short (*srid)[36]    = (short (*)[36])sbuf;  // 32*36*2  = 2304 B

  const int tid = threadIdx.x;
  const int blk = blockIdx.x;
  const int n   = blk >> 7;          // batch
  const int rt  = blk & 127;
  const int tq  = rt >> 5;           // query level
  const int h   = rt & 31;           // query grid row

  // ---- stage query neighborhood rows h-1..h+1 (zero-padded), f32 -> bf16 ----
  {
    const int ch = tid;
    for (int j = 0; j < 3 * 34; ++j) {
      int dh  = j / 34;
      int w34 = j - dh * 34;
      int col = w34 - 1;
      int hh  = h + dh - 1;
      float v = 0.f;
      if ((unsigned)col < 32u && (unsigned)hh < 32u)
        v = query[(size_t)(n * LEN + tq * 1024 + hh * 32 + col) * DD + ch];
      Qs[dh][w34][ch] = (__bf16)v;
    }
  }

  // ---- zero Vs pad columns (0, 33) once; async copies never touch them ----
  for (int i = tid; i < 4 * 2 * 128; i += 128) {
    int slab = i >> 8, side = (i >> 7) & 1, ch = i & 127;
    Vs[slab][side ? 33 : 0][ch] = (__bf16)0.f;
  }

  // slab helpers: one grid row = 32*128 bf16 = 8192 B, contiguous both sides
  auto zero_slab = [&](int slab) {
    uint4* dst = (uint4*)&Vs[slab][1][0];
    for (int c = tid; c < 512; c += 128) dst[c] = make_uint4(0u, 0u, 0u, 0u);
  };
  auto stage_async = [&](int slab, int row, const __bf16* lvl) {
    uint32_t lbase = (uint32_t)(uintptr_t)&Vs[slab][1][0];
    uint64_t g = (uint64_t)(uintptr_t)(lvl + (size_t)row * 32 * DD);
    for (int c = tid; c < 512; c += 128) {       // 4 iters, uniform EXEC
      uint32_t laddr = lbase + (uint32_t)c * 16u;
      uint32_t goff  = (uint32_t)c * 16u;
      asm volatile("global_load_async_to_lds_b128 %0, %1, %2"
                   :: "v"(laddr), "v"(goff), "s"(g) : "memory");
    }
  };

  const int wave = tid >> 5, lane = tid & 31;
  const int m0 = (wave & 1) << 4;    // query sub-tile
  const int n0 = (wave >> 1) << 4;   // key sub-tile
  const int l  = lane & 15, hiv = lane >> 4;

  float acc[32];
  for (int q = 0; q < 32; ++q) acc[q] = 0.f;

  for (int tk = 0; tk < TT; ++tk) {
    const __bf16* lvl = Vb + (size_t)(n * LEN + tk * 1024) * DD;

    // prologue: ring holds rows -1 (zeros), 0, 1
    zero_slab(3);
    stage_async(0, 0, lvl);
    stage_async(1, 1, lvl);
    asm volatile("s_wait_asynccnt 0x0" ::: "memory");
    __syncthreads();

    float tv0 = -1e30f, tv1 = -1e30f, tv2 = -1e30f, tv3 = -1e30f;
    int   ti0 = 0, ti1 = 0, ti2 = 0, ti3 = 0;

    for (int kh = 0; kh < 32; ++kh) {
      // ---- prefetch row kh+2 into the retired slab (overlaps WMMA) ----
      {
        int nr = kh + 2;
        if (nr < 32)       stage_async(nr & 3, nr, lvl);
        else if (nr == 32) zero_slab(0);         // row 32 -> zeros (slab 0)
      }

      // ---- 32x32 corr tile: sum over 9 shifts x 128 ch via bf16 WMMA ----
      v8f c = {};
      for (int kk = 0; kk < 9; ++kk) {
        const int dh = kk / 3;
        const int dw = kk % 3 - 1;
        const __bf16* qp = &Qs[dh][m0 + l + dw + 1][hiv * 8];
        const __bf16* vp = &Vs[(kh + dh + 3) & 3][n0 + l + dw + 1][hiv * 8];
        #pragma unroll
        for (int kc = 0; kc < 128; kc += 32) {
          v16bf a, b;
          #pragma unroll
          for (int j = 0; j < 8; ++j) {
            a[j]     = qp[kc + j];        // K = hiv*8 + j
            a[j + 8] = qp[kc + 16 + j];   // K = 16 + hiv*8 + j
          }
          #pragma unroll
          for (int j = 0; j < 8; ++j) {
            b[j]     = vp[kc + j];        // K = hiv*8 + j, column N = l
            b[j + 8] = vp[kc + 16 + j];   // K = 16 + hiv*8 + j
          }
          c = __builtin_amdgcn_wmma_f32_16x16x32_bf16(
                false, a, false, b, (short)0, c, false, false);
        }
      }
      #pragma unroll
      for (int r = 0; r < 8; ++r)
        corrbuf[m0 + hiv * 8 + r][n0 + l] = c[r] * 0.0078125f;  // /dim

      asm volatile("s_wait_asynccnt 0x0" ::: "memory");  // drain prefetch
      __syncthreads();

      // ---- merge 32 new corr values into running top-4 ----
      if (tid < 32) {
        for (int k2 = 0; k2 < 32; ++k2) {
          float v  = corrbuf[tid][k2];
          int pidx = kh * 32 + k2;
          if (v > tv3) {
            if (v > tv0)      { tv3=tv2; ti3=ti2; tv2=tv1; ti2=ti1; tv1=tv0; ti1=ti0; tv0=v; ti0=pidx; }
            else if (v > tv1) { tv3=tv2; ti3=ti2; tv2=tv1; ti2=ti1; tv1=v; ti1=pidx; }
            else if (v > tv2) { tv3=tv2; ti3=ti2; tv2=v; ti2=pidx; }
            else              { tv3=v; ti3=pidx; }
          }
        }
      }
      __syncthreads();
    } // kh

    // ---- indices -> sampling_locations + shared ridx ----
    if (tid < 32) {
      const int qg = tq * 1024 + h * 32 + tid;
      const int shifts[9] = {-33, -32, -31, -1, 0, 1, 31, 32, 33};
      int tis[4] = {ti0, ti1, ti2, ti3};
      #pragma unroll
      for (int kk = 0; kk < 9; ++kk) {
        #pragma unroll
        for (int np = 0; np < 4; ++np) {
          int p = kk * 4 + np;                 // reshape order: (9, NP)
          int r = tis[np] + shifts[kk];
          r = r < 0 ? 0 : (r > 961 ? 961 : r); // clip to (Hs-1)*(Ws-1)
          srid[tid][p] = (short)r;
          int ow = r >> 5, oh = r & 31;
          size_t sb = (size_t)OUT_OFF +
                      ((((size_t)(n * LEN + qg) * TT + tk) * P9A + p) << 1);
          out[sb]     = (float)ow * 0.03125f;
          out[sb + 1] = (float)oh * 0.03125f;
        }
      }
    }
    __syncthreads();

    // ---- gather-average accumulate (thread = channel d), exact f32 data ----
    {
      const int d = tid;
      const float wc = 0.25f / 144.0f;
      const float* vbase = vflat + (size_t)(n * LEN + tk * 1024) * DD + d;
      for (int q = 0; q < 32; ++q) {
        float a = acc[q];
        for (int p = 0; p < 36; ++p) {
          int r  = (int)srid[q][p];
          int ow = r >> 5, oh = r & 31;
          #pragma unroll
          for (int dy = 0; dy < 2; ++dy) {
            int yi = oh - 1 + dy;
            #pragma unroll
            for (int dx = 0; dx < 2; ++dx) {
              int xi = ow - 1 + dx;
              if ((unsigned)xi < 32u && (unsigned)yi < 32u)
                a += wc * vbase[(size_t)(yi * 32 + xi) * DD];
            }
          }
        }
        acc[q] = a;
      }
    }
    __syncthreads();  // srid consumed before next level reuses sbuf
  } // tk

  // ---- output projection: out = acc @ W_out^T + b ----
  float* accbuf = (float*)&Qs[0][0][0];        // 16 KB <= sizeof(Qs)
  for (int q = 0; q < 32; ++q) accbuf[q * 128 + tid] = acc[q];
  __syncthreads();
  {
    const int doch = tid;
    const float bb = bout[doch];
    const float* wrow = Wout + (size_t)doch * 128;
    const int qgb = tq * 1024 + h * 32;
    for (int q = 0; q < 32; ++q) {
      float s = bb;
      for (int di = 0; di < 128; ++di)
        s += accbuf[q * 128 + di] * wrow[di];
      out[(size_t)(n * LEN + qgb + q) * DD + doch] = s;
    }
  }
}

// ---------------------------------------------------------------------------
// Fallback (round-1): synchronous f32 staging, no workspace
// ---------------------------------------------------------------------------
__launch_bounds__(128)
__global__ void msdma_fused(const float* __restrict__ query,
                            const float* __restrict__ vflat,
                            const float* __restrict__ Wout,
                            const float* __restrict__ bout,
                            float* __restrict__ out)
{
  __shared__ __bf16 Qs[3][34][128];
  __shared__ __bf16 Vs[3][34][128];
  __shared__ float  corrbuf[32][32];
  __shared__ int    sridx[32][36];

  const int tid = threadIdx.x;
  const int blk = blockIdx.x;
  const int n   = blk >> 7;
  const int rt  = blk & 127;
  const int tq  = rt >> 5;
  const int h   = rt & 31;

  {
    const int ch = tid;
    for (int j = 0; j < 3 * 34; ++j) {
      int dh = j / 34, w34 = j - dh * 34, col = w34 - 1, hh = h + dh - 1;
      float v = 0.f;
      if ((unsigned)col < 32u && (unsigned)hh < 32u)
        v = query[(size_t)(n * LEN + tq * 1024 + hh * 32 + col) * DD + ch];
      Qs[dh][w34][ch] = (__bf16)v;
    }
  }

  const int wave = tid >> 5, lane = tid & 31;
  const int m0 = (wave & 1) << 4, n0 = (wave >> 1) << 4;
  const int l = lane & 15, hiv = lane >> 4;

  float acc[32];
  for (int q = 0; q < 32; ++q) acc[q] = 0.f;

  for (int tk = 0; tk < TT; ++tk) {
    float tv0 = -1e30f, tv1 = -1e30f, tv2 = -1e30f, tv3 = -1e30f;
    int   ti0 = 0, ti1 = 0, ti2 = 0, ti3 = 0;

    for (int kh = 0; kh < 32; ++kh) {
      __syncthreads();
      {
        const int ch = tid;
        for (int j = 0; j < 3 * 34; ++j) {
          int dh = j / 34, w34 = j - dh * 34, col = w34 - 1, hh = kh + dh - 1;
          float v = 0.f;
          if ((unsigned)col < 32u && (unsigned)hh < 32u)
            v = vflat[(size_t)(n * LEN + tk * 1024 + hh * 32 + col) * DD + ch];
          Vs[dh][w34][ch] = (__bf16)v;
        }
      }
      __syncthreads();

      v8f c = {};
      for (int kk = 0; kk < 9; ++kk) {
        const int dh = kk / 3;
        const int dw = kk % 3 - 1;
        const __bf16* qp = &Qs[dh][m0 + l + dw + 1][hiv * 8];
        const __bf16* vp = &Vs[dh][n0 + l + dw + 1][hiv * 8];
        #pragma unroll
        for (int kc = 0; kc < 128; kc += 32) {
          v16bf a, b;
          #pragma unroll
          for (int j = 0; j < 8; ++j) {
            a[j]     = qp[kc + j];
            a[j + 8] = qp[kc + 16 + j];
          }
          #pragma unroll
          for (int j = 0; j < 8; ++j) { b[j] = vp[kc + j]; b[j + 8] = vp[kc + 16 + j]; }
          c = __builtin_amdgcn_wmma_f32_16x16x32_bf16(
                false, a, false, b, (short)0, c, false, false);
        }
      }
      #pragma unroll
      for (int r = 0; r < 8; ++r)
        corrbuf[m0 + hiv * 8 + r][n0 + l] = c[r] * 0.0078125f;
      __syncthreads();

      if (tid < 32) {
        for (int k2 = 0; k2 < 32; ++k2) {
          float v = corrbuf[tid][k2];
          int pidx = kh * 32 + k2;
          if (v > tv3) {
            if (v > tv0)      { tv3=tv2; ti3=ti2; tv2=tv1; ti2=ti1; tv1=tv0; ti1=ti0; tv0=v; ti0=pidx; }
            else if (v > tv1) { tv3=tv2; ti3=ti2; tv2=tv1; ti2=ti1; tv1=v; ti1=pidx; }
            else if (v > tv2) { tv3=tv2; ti3=ti2; tv2=v; ti2=pidx; }
            else              { tv3=v; ti3=pidx; }
          }
        }
      }
    }

    if (tid < 32) {
      const int qg = tq * 1024 + h * 32 + tid;
      const int shifts[9] = {-33, -32, -31, -1, 0, 1, 31, 32, 33};
      int tis[4] = {ti0, ti1, ti2, ti3};
      #pragma unroll
      for (int kk = 0; kk < 9; ++kk) {
        #pragma unroll
        for (int np = 0; np < 4; ++np) {
          int p = kk * 4 + np;
          int r = tis[np] + shifts[kk];
          r = r < 0 ? 0 : (r > 961 ? 961 : r);
          sridx[tid][p] = r;
          int ow = r >> 5, oh = r & 31;
          size_t sb = (size_t)OUT_OFF +
                      ((((size_t)(n * LEN + qg) * TT + tk) * P9A + p) << 1);
          out[sb]     = (float)ow * 0.03125f;
          out[sb + 1] = (float)oh * 0.03125f;
        }
      }
    }
    __syncthreads();

    {
      const int d = tid;
      const float wc = 0.25f / 144.0f;
      const float* vbase = vflat + (size_t)(n * LEN + tk * 1024) * DD + d;
      for (int q = 0; q < 32; ++q) {
        float a = acc[q];
        for (int p = 0; p < 36; ++p) {
          int r = sridx[q][p];
          int ow = r >> 5, oh = r & 31;
          #pragma unroll
          for (int dy = 0; dy < 2; ++dy) {
            int yi = oh - 1 + dy;
            #pragma unroll
            for (int dx = 0; dx < 2; ++dx) {
              int xi = ow - 1 + dx;
              if ((unsigned)xi < 32u && (unsigned)yi < 32u)
                a += wc * vbase[(size_t)(yi * 32 + xi) * DD];
            }
          }
        }
        acc[q] = a;
      }
    }
    __syncthreads();
  }

  float* accbuf = (float*)&Qs[0][0][0];
  for (int q = 0; q < 32; ++q) accbuf[q * 128 + tid] = acc[q];
  __syncthreads();
  {
    const int doch = tid;
    const float bb = bout[doch];
    const float* wrow = Wout + (size_t)doch * 128;
    const int qgb = tq * 1024 + h * 32;
    for (int q = 0; q < 32; ++q) {
      float s = bb;
      for (int di = 0; di < 128; ++di)
        s += accbuf[q * 128 + di] * wrow[di];
      out[(size_t)(n * LEN + qgb + q) * DD + doch] = s;
    }
  }
}

extern "C" void kernel_launch(void* const* d_in, const int* in_sizes, int n_in,
                              void* d_out, int out_size, void* d_ws, size_t ws_size,
                              hipStream_t stream) {
  (void)in_sizes; (void)n_in; (void)out_size;
  const float* query = (const float*)d_in[0];
  const float* vflat = (const float*)d_in[2];   // input_flatten
  const float* Wout  = (const float*)d_in[5];
  const float* bout  = (const float*)d_in[6];
  float* out = (float*)d_out;

  const int ntot = NB * LEN * DD;               // 1,048,576
  dim3 grid(NB * TT * HS);                      // 256 blocks
  dim3 block(128);                              // 4 wave32

  if (ws_size >= (size_t)ntot * sizeof(__bf16)) {
    __bf16* Vb = (__bf16*)d_ws;
    cvt_bf16<<<dim3((ntot + 255) / 256), dim3(256), 0, stream>>>(vflat, Vb, ntot);
    msdma_fused_async<<<grid, block, 0, stream>>>(query, vflat, Vb, Wout, bout, out);
  } else {
    msdma_fused<<<grid, block, 0, stream>>>(query, vflat, Wout, bout, out);
  }
}
